// Generator_45586782880060
// MI455X (gfx1250) — matrix operand order repaired and checked by reading, not compiled, not run on hardware
//
#include <hip/hip_runtime.h>

// ---------------------------------------------------------------------------
// Types for CDNA5 WMMA (gfx1250, wave32)
// ---------------------------------------------------------------------------
typedef __attribute__((ext_vector_type(16))) __bf16 v16bf;
typedef __attribute__((ext_vector_type(8)))  __bf16 v8bf;
typedef __attribute__((ext_vector_type(8)))  float  v8f;

// Explicit global-address-space views so H traffic compiles to
// global_load/global_store (FLAT would also tie up DScnt + LDS datapath).
typedef const __attribute__((address_space(1))) v8bf*   gv8bf_c;
typedef const __attribute__((address_space(1))) v16bf*  gv16bf_c;
typedef __attribute__((address_space(1))) __bf16*       gbf;
typedef __attribute__((address_space(1))) float*        gf32;

#define Bsz      256
#define HID      512
#define SEQ      64
#define TSTEPS   64
#define NLAYERS  4
#define NBLOCKS  32                 // 32 n-tiles; each block covers all 256 rows
#define NPHASES  (TSTEPS * NLAYERS) // one barrier per layer-phase
#define LDS_ROWB 1040               // bytes per LDS weight row (512*2 + 16 pad)
#define LDS_BYTES ((192 + 16) * LDS_ROWB) // 4 layers*3 gates*16 rows + Wout 16 rows

__device__ __forceinline__ __bf16 f2bf(float f) { return (__bf16)f; }

__device__ __forceinline__ v16bf cat16(v8bf lo, v8bf hi) {
  return __builtin_shufflevector(lo, hi, 0,1,2,3,4,5,6,7,8,9,10,11,12,13,14,15);
}

// CDNA5 hardware v_tanh_f32 (TRANS op) when available.
#if __has_builtin(__builtin_amdgcn_tanhf)
__device__ __forceinline__ float fast_tanh(float x) {
  return __builtin_amdgcn_tanhf(x);
}
#else
__device__ __forceinline__ float fast_tanh(float x) {
  float e = __expf(2.0f * x);
  return (e - 1.0f) * __builtin_amdgcn_rcpf(e + 1.0f);
}
#endif
__device__ __forceinline__ float fast_sigmoid(float x) {
  return 0.5f * fast_tanh(0.5f * x) + 0.5f;
}

// Agent-scope grid barrier; one fresh counter per phase (no sense reversal).
__device__ __forceinline__ void grid_barrier(int* cnt, int phase) {
  __syncthreads();
  if (threadIdx.x == 0) {
    __hip_atomic_fetch_add(cnt + phase, 1, __ATOMIC_ACQ_REL, __HIP_MEMORY_SCOPE_AGENT);
    while (__hip_atomic_load(cnt + phase, __ATOMIC_ACQUIRE, __HIP_MEMORY_SCOPE_AGENT) < NBLOCKS)
      __builtin_amdgcn_s_sleep(1);
  }
  __syncthreads();
}

// ---------------------------------------------------------------------------
// Kernel 1: weight prep + barrier-counter reset (runs every call).
// ---------------------------------------------------------------------------
__global__ void __launch_bounds__(256)
prep_weights(const float* __restrict__ W_ih,
             const float* __restrict__ b_ih,
             const float* __restrict__ b_hh,
             const float* __restrict__ W_out,
             __bf16* __restrict__ Wih_bf,
             float*  __restrict__ bsum,
             __bf16* __restrict__ Wout_bf,
             int*    __restrict__ cnt) {
  const int stride = gridDim.x * blockDim.x;
  const int tid = blockIdx.x * blockDim.x + threadIdx.x;
  for (int j = tid; j < NLAYERS * 4 * HID * HID; j += stride)
    Wih_bf[j] = f2bf(W_ih[j]);
  for (int j = tid; j < SEQ * HID; j += stride)
    Wout_bf[j] = f2bf(W_out[j]);
  for (int j = tid; j < NLAYERS * 4 * HID; j += stride)
    bsum[j] = b_ih[j] + b_hh[j];
  for (int j = tid; j < NPHASES; j += stride)
    cnt[j] = 0;
}

// ---------------------------------------------------------------------------
// Kernel 2: head part 1 — argmax expert, one-hot column gather (ln), noise GEMV.
// ---------------------------------------------------------------------------
__global__ void __launch_bounds__(128)
head1(const float* __restrict__ label,
      const float* __restrict__ noise,
      const int*   __restrict__ length,
      const float* __restrict__ W_len,
      const float* __restrict__ b_len,
      const float* __restrict__ W_noise,
      const float* __restrict__ b_noise,
      float* __restrict__ comb) {
  const int b = blockIdx.x;
  const int tid = threadIdx.x;   // 0..127
  int e = 0; float best = label[b * 8];
  #pragma unroll
  for (int j = 1; j < 8; ++j) {
    float v = label[b * 8 + j];
    if (v > best) { best = v; e = j; }
  }
  const int col = length[b] - 1;
  float ln = W_len[(e * 128 + tid) * 64 + col] + b_len[e * 128 + tid];
  comb[b * 256 + tid] = ln > 0.f ? ln : 0.f;
  float acc = b_noise[e * 128 + tid];
  const float* w = W_noise + (e * 128 + tid) * 128;
  const float* x = noise + b * 128;
  #pragma unroll 4
  for (int k = 0; k < 128; ++k) acc += w[k] * x[k];
  comb[b * 256 + 128 + tid] = acc > 0.f ? acc : 0.f;
}

// ---------------------------------------------------------------------------
// Kernel 3: head part 2 — x0 = relu(W_tail[idx] @ comb + b_tail[idx]) -> bf16 H0
// ---------------------------------------------------------------------------
__global__ void __launch_bounds__(256)
head2(const float* __restrict__ label,
      const float* __restrict__ W_tail,
      const float* __restrict__ b_tail,
      const float* __restrict__ comb,
      __bf16* __restrict__ H0) {
  __shared__ float xs[256];
  const int b = blockIdx.x;
  const int tid = threadIdx.x;
  xs[tid] = comb[b * 256 + tid];
  int e = 0; float best = label[b * 8];
  #pragma unroll
  for (int j = 1; j < 8; ++j) {
    float v = label[b * 8 + j];
    if (v > best) { best = v; e = j; }
  }
  __syncthreads();
  #pragma unroll
  for (int o = tid; o < HID; o += 256) {
    float acc = b_tail[e * HID + o];
    const float* w = W_tail + (e * HID + o) * 256;
    #pragma unroll 4
    for (int k = 0; k < 256; ++k) acc += w[k] * xs[k];
    acc = acc > 0.f ? acc : 0.f;
    H0[b * HID + o] = f2bf(acc);
  }
}

// ---------------------------------------------------------------------------
// Kernel 4: persistent recurrence kernel.
// Grid: 32 blocks (one n-tile each), 256 threads = 8 waves; wave w covers rows
// w*32..w*32+31 (all 256 batch rows per block => 2 waves per SIMD32 for
// latency hiding). Each block preloads its fixed B-operand slice for all 4
// layers (3 live gates x 16 rows x 512 k, bf16, padded rows) = 192 KB into
// LDS, plus 16 KB of W_out for blocks nx<4; then iterates 64 steps x 4 layers
// with an agent-scope barrier per layer. h ping-pongs through L2 (explicit
// addrspace(1) access => global_load/store); B comes from LDS (ds_load_b128).
//
// WMMA layouts used by this kernel:
//   A 16x32 bf16: lane row = lane%16; K = k0 + half*16 .. +15 (contiguous)
//   B 32x16 bf16: lane col = lane%16; K = k0 + half*16 .. +15 (contiguous)
//   C/D f32:      elem r -> (M = r + 8*half, N = lane%16)
// ---------------------------------------------------------------------------
__global__ void __launch_bounds__(256)
lstm_persistent(const __bf16* __restrict__ Wih_bf,
                const float*  __restrict__ bsum,
                const __bf16* __restrict__ Wout_bf,
                const float*  __restrict__ b_out,
                __bf16* __restrict__ H0,
                __bf16* __restrict__ H1,
                float*  __restrict__ out,
                int*    __restrict__ cnt) {
  extern __shared__ char smem[];
  const int tid  = threadIdx.x;
  const int wave = tid >> 5;              // 0..7
  const int lane = tid & 31;
  const int half = lane >> 4;
  const int lr   = lane & 15;
  const int nx   = blockIdx.x;            // n-tile 0..31
  const int n0   = nx * 16;
  const int m0   = wave * 32;             // 8 waves x 32 rows = 256 rows

  // ---- Preload weights into LDS (one-time; weights read from HBM once) ----
  // LDS row index: (l*3 + q)*16 + r  for layer l, gate q in {i,g,o}, row r.
  #pragma unroll
  for (int l = 0; l < NLAYERS; ++l) {
    #pragma unroll
    for (int q = 0; q < 3; ++q) {
      const int goff = (q == 0) ? 0 : (q == 1 ? 1024 : 1536);
      const __bf16* src = Wih_bf + ((size_t)l * 4 * HID + goff + n0) * HID;
      char* dst = smem + (size_t)((l * 3 + q) * 16) * LDS_ROWB;
      for (int i = tid; i < 16 * 64; i += 256) {         // 16 rows x 64 x 16B
        const int r = i >> 6, c = i & 63;
        *(uint4*)(dst + r * LDS_ROWB + c * 16) =
            *(const uint4*)((const char*)(src + (size_t)r * HID) + c * 16);
      }
    }
  }
  if (nx < 4) {  // W_out rows n0..n0+15 -> LDS rows 192..207
    const __bf16* src = Wout_bf + (size_t)n0 * HID;
    char* dst = smem + (size_t)192 * LDS_ROWB;
    for (int i = tid; i < 16 * 64; i += 256) {
      const int r = i >> 6, c = i & 63;
      *(uint4*)(dst + r * LDS_ROWB + c * 16) =
          *(const uint4*)((const char*)(src + (size_t)r * HID) + c * 16);
    }
  }
  __syncthreads();

  // Per-lane biases (column n0+lr), all layers, preloaded into registers.
  const int col = n0 + lr;
  float bi[NLAYERS], bg[NLAYERS], bo[NLAYERS];
  #pragma unroll
  for (int l = 0; l < NLAYERS; ++l) {
    bi[l] = bsum[l * 4 * HID + col];
    bg[l] = bsum[l * 4 * HID + 1024 + col];
    bo[l] = bsum[l * 4 * HID + 1536 + col];
  }
  const float bv = b_out[(n0 & 63) + lr];  // only meaningful for nx<4

  const __bf16* Hc = H0;   // explicit swap (no alloca => addrspace inference)
  __bf16*       Hw = H1;
  int phase = 0;

  for (int t = 0; t < TSTEPS; ++t) {
    for (int l = 0; l < NLAYERS; ++l) {
      const char* sm_i = smem + (size_t)((l * 3 + 0) * 16 + lr) * LDS_ROWB + half * 32;
      const char* sm_g = smem + (size_t)((l * 3 + 1) * 16 + lr) * LDS_ROWB + half * 32;
      const char* sm_o = smem + (size_t)((l * 3 + 2) * 16 + lr) * LDS_ROWB + half * 32;
      const __bf16* hr0 = Hc + (size_t)(m0 + lr)      * HID;
      const __bf16* hr1 = Hc + (size_t)(m0 + 16 + lr) * HID;

      v8f ai0 = {}, ai1 = {}, ag0 = {}, ag1 = {}, ao0 = {}, ao1 = {};
      #pragma unroll 4
      for (int k0 = 0; k0 < HID; k0 += 32) {
        v16bf A0 = cat16(*(gv8bf_c)(hr0 + k0 + half * 16),
                         *(gv8bf_c)(hr0 + k0 + 8 + half * 16));
        v16bf A1 = cat16(*(gv8bf_c)(hr1 + k0 + half * 16),
                         *(gv8bf_c)(hr1 + k0 + 8 + half * 16));
        v16bf Bi = *(const v16bf*)(sm_i + k0 * 2);
        v16bf Bg = *(const v16bf*)(sm_g + k0 * 2);
        v16bf Bo = *(const v16bf*)(sm_o + k0 * 2);
        ai0 = __builtin_amdgcn_wmma_f32_16x16x32_bf16(false, A0, false, Bi, (short)0, ai0, false, false);
        ai1 = __builtin_amdgcn_wmma_f32_16x16x32_bf16(false, A1, false, Bi, (short)0, ai1, false, false);
        ag0 = __builtin_amdgcn_wmma_f32_16x16x32_bf16(false, A0, false, Bg, (short)0, ag0, false, false);
        ag1 = __builtin_amdgcn_wmma_f32_16x16x32_bf16(false, A1, false, Bg, (short)0, ag1, false, false);
        ao0 = __builtin_amdgcn_wmma_f32_16x16x32_bf16(false, A0, false, Bo, (short)0, ao0, false, false);
        ao1 = __builtin_amdgcn_wmma_f32_16x16x32_bf16(false, A1, false, Bo, (short)0, ao1, false, false);
      }

      __bf16 res0[8], res1[8];
      #pragma unroll
      for (int r = 0; r < 8; ++r) {
        float c0 = fast_sigmoid(ai0[r] + bi[l]) * fast_tanh(ag0[r] + bg[l]);
        float c1 = fast_sigmoid(ai1[r] + bi[l]) * fast_tanh(ag1[r] + bg[l]);
        res0[r] = f2bf(fast_sigmoid(ao0[r] + bo[l]) * fast_tanh(c0));
        res1[r] = f2bf(fast_sigmoid(ao1[r] + bo[l]) * fast_tanh(c1));
      }
      #pragma unroll
      for (int r = 0; r < 8; ++r) {
        const int row0 = m0 + r + 8 * half;
        *((gbf)(Hw + (size_t)row0 * HID + col))        = res0[r];
        *((gbf)(Hw + (size_t)(row0 + 16) * HID + col)) = res1[r];
      }
      { const __bf16* tmp = Hc; Hc = Hw; Hw = (__bf16*)tmp; }
      grid_barrier(cnt, phase++);
    }

    // Output projection: h (= Hc = H0 after 4 swaps) -> out[:, t, :].
    // Safe to overlap with next step's layer 0: Hc isn't overwritten until
    // after the *next* barrier, and every block finishes these stores before
    // signaling it.
    if (nx < 4) {
      const char*   smw = smem + (size_t)(192 + lr) * LDS_ROWB + half * 32;
      const __bf16* hr0 = Hc + (size_t)(m0 + lr)      * HID;
      const __bf16* hr1 = Hc + (size_t)(m0 + 16 + lr) * HID;
      v8f a0 = {}, a1 = {};
      #pragma unroll 4
      for (int k0 = 0; k0 < HID; k0 += 32) {
        v16bf A0 = cat16(*(gv8bf_c)(hr0 + k0 + half * 16),
                         *(gv8bf_c)(hr0 + k0 + 8 + half * 16));
        v16bf A1 = cat16(*(gv8bf_c)(hr1 + k0 + half * 16),
                         *(gv8bf_c)(hr1 + k0 + 8 + half * 16));
        v16bf B  = *(const v16bf*)(smw + k0 * 2);
        a0 = __builtin_amdgcn_wmma_f32_16x16x32_bf16(false, A0, false, B, (short)0, a0, false, false);
        a1 = __builtin_amdgcn_wmma_f32_16x16x32_bf16(false, A1, false, B, (short)0, a1, false, false);
      }
      #pragma unroll
      for (int r = 0; r < 8; ++r) {
        const int row0 = m0 + r + 8 * half;
        const int row1 = row0 + 16;
        *((gf32)(out + (size_t)row0 * (TSTEPS * SEQ) + t * SEQ + col)) = fast_tanh(a0[r] + bv);
        *((gf32)(out + (size_t)row1 * (TSTEPS * SEQ) + t * SEQ + col)) = fast_tanh(a1[r] + bv);
      }
    }
  }
}

// ---------------------------------------------------------------------------
// Launch
// ---------------------------------------------------------------------------
extern "C" void kernel_launch(void* const* d_in, const int* in_sizes, int n_in,
                              void* d_out, int out_size, void* d_ws, size_t ws_size,
                              hipStream_t stream) {
  const float* label   = (const float*)d_in[0];
  const float* noise   = (const float*)d_in[1];
  const int*   length  = (const int*)  d_in[2];
  const float* W_len   = (const float*)d_in[3];
  const float* b_len   = (const float*)d_in[4];
  const float* W_noise = (const float*)d_in[5];
  const float* b_noise = (const float*)d_in[6];
  const float* W_tail  = (const float*)d_in[7];
  const float* b_tail  = (const float*)d_in[8];
  const float* W_ih    = (const float*)d_in[9];
  const float* b_ih    = (const float*)d_in[10];
  // W_hh (d_in[11]) only contributes through its bias partner:
  const float* b_hh    = (const float*)d_in[12];
  const float* W_out   = (const float*)d_in[13];
  const float* b_out   = (const float*)d_in[14];
  float* out = (float*)d_out;

  // Workspace layout (bytes)
  char* ws = (char*)d_ws;
  __bf16* Wih_bf  = (__bf16*)(ws + 0);                    // 4*2048*512*2 = 8388608
  float*  bsum    = (float*) (ws + 8388608);              // 4*2048*4     = 32768
  __bf16* Wout_bf = (__bf16*)(ws + 8421376);              // 64*512*2     = 65536
  float*  comb    = (float*) (ws + 8486912);              // 256*256*4    = 262144
  __bf16* H0      = (__bf16*)(ws + 8749056);              // 256*512*2    = 262144
  __bf16* H1      = (__bf16*)(ws + 9011200);              // 256*512*2    = 262144
  int*    cnt     = (int*)   (ws + 9273344);              // 256*4        = 1024

  prep_weights<<<2048, 256, 0, stream>>>(W_ih, b_ih, b_hh, W_out,
                                         Wih_bf, bsum, Wout_bf, cnt);
  head1<<<Bsz, 128, 0, stream>>>(label, noise, length, W_len, b_len,
                                 W_noise, b_noise, comb);
  head2<<<Bsz, 256, 0, stream>>>(label, W_tail, b_tail, comb, H0);

  lstm_persistent<<<32, 256, LDS_BYTES, stream>>>(
      Wih_bf, bsum, Wout_bf, b_out, H0, H1, out, cnt);
}
